// MultiHeadAttention_80942953660675
// MI455X (gfx1250) — matrix operand order repaired and checked by reading, not compiled, not run on hardware
//
#include <hip/hip_runtime.h>

// ---------------------------------------------------------------------------
// MHA forward for MI455X (gfx1250, wave32, WMMA).
// Stage 0: fp32 -> bf16 conversion of activations+weights (one pass).
// Stages 1-3: all GEMMs on v_wmma_f32_16x16x32_bf16 (f32 accum).
// Shared operands (weights / K / V) are staged through LDS with double
// buffering so the 8 waves of a block fetch them from L2 only once, and the
// global->LDS fill of step k+1 overlaps the WMMA work of step k.
// ---------------------------------------------------------------------------

#define DEV __device__ __forceinline__

typedef __bf16 bf16;
typedef __attribute__((ext_vector_type(16))) __bf16 v16bf;
typedef __attribute__((ext_vector_type(8)))  __bf16 bf16x8;
typedef __attribute__((ext_vector_type(4)))  __bf16 bf16x4;
typedef __attribute__((ext_vector_type(4)))  float  f32x4;
typedef __attribute__((ext_vector_type(8)))  float  v8f;

static constexpr int Bc  = 2;
static constexpr int Tc  = 2048;
static constexpr int Dc  = 1024;
static constexpr int Hc  = 16;
static constexpr int HDc = 64;   // head dim

// padded LDS leading dims (bank-conflict avoidance: 40*2B=80B, 72*2B=144B)
static constexpr int LDB = 40;   // for 32-wide K-slices
static constexpr int LDK = 72;   // for 64-wide K rows

// ---------------------------------------------------------------------------
// WMMA wrapper: D = A(16x32 bf16) * B(32x16 bf16) + C(16x16 f32)
// ---------------------------------------------------------------------------
DEV v8f wmma_bf16(v16bf a, v16bf b, v8f c) {
  return __builtin_amdgcn_wmma_f32_16x16x32_bf16(
      /*neg_a=*/false, a, /*neg_b=*/false, b,
      /*c_mod=*/(short)0, c, /*reuse_a=*/false, /*reuse_b=*/false);
}

// ---------------------------------------------------------------------------
// Fragment loaders (layouts per CDNA5 ISA 7.12.2):
//  A 16x32 bf16: lane -> row = lane%16; half h: K = (h/8)*16 + (lane/16)*8 + h%8
//  B 32x16 bf16: lane -> col = lane%16; half h: K = (h/8)*16 + (lane/16)*8 + h%8
//  C 16x16 f32 : lane -> col = lane%16; vgpr r: row = r + (lane/16)*8
// Sources are "row-major along K"; work for both global and LDS pointers.
// ---------------------------------------------------------------------------
DEV v16bf load_A_frag(const bf16* src, int ld) {
  const int lane = threadIdx.x & 31;
  const bf16* p = src + (lane & 15) * ld + ((lane >> 4) << 3);
  union { v16bf v; bf16x8 h[2]; } u;
  u.h[0] = *reinterpret_cast<const bf16x8*>(p);
  u.h[1] = *reinterpret_cast<const bf16x8*>(p + 16);
  return u.v;
}

DEV v16bf load_B_frag(const bf16* srcBt, int ld) {
  // srcBt is B^T, i.e. srcBt[n*ld + k] == B[k][n]
  const int lane = threadIdx.x & 31;
  const bf16* p = srcBt + (lane & 15) * ld + ((lane >> 4) << 3);
  union { v16bf v; bf16x8 h[2]; } u;
  u.h[0] = *reinterpret_cast<const bf16x8*>(p);
  u.h[1] = *reinterpret_cast<const bf16x8*>(p + 16);
  return u.v;
}

DEV void copy16(bf16* dst, const bf16* src) {  // 2 x b128
  *reinterpret_cast<bf16x8*>(dst)     = *reinterpret_cast<const bf16x8*>(src);
  *reinterpret_cast<bf16x8*>(dst + 8) = *reinterpret_cast<const bf16x8*>(src + 8);
}
DEV void copy8(bf16* dst, const bf16* src) {   // 1 x b128
  *reinterpret_cast<bf16x8*>(dst) = *reinterpret_cast<const bf16x8*>(src);
}

// ---------------------------------------------------------------------------
// Stage 0: elementwise fp32 -> bf16 (vectorized x4).
// ---------------------------------------------------------------------------
__global__ __launch_bounds__(256) void cvt_bf16_kernel(
    const float* __restrict__ src, bf16* __restrict__ dst, int n4) {
  const int i = blockIdx.x * blockDim.x + threadIdx.x;
  if (i < n4) {
    f32x4 v = *reinterpret_cast<const f32x4*>(src + (size_t)i * 4);
    bf16x4 o;
#pragma unroll
    for (int j = 0; j < 4; ++j) o[j] = (bf16)v[j];
    *reinterpret_cast<bf16x4*>(dst + (size_t)i * 4) = o;
  }
}

// ---------------------------------------------------------------------------
// Stage 1: fused QKV projection.  y = x @ W^T  (NT GEMM), all-bf16 operands.
// Block task: one weight, 8 M-tiles (one per wave) x 128 N cols (shared).
// B slice (128 x 32) staged in LDS, double-buffered.
// Q,K stored [B,H,T,hd]; V stored transposed [B,H,hd,T].
// ---------------------------------------------------------------------------
__global__ __launch_bounds__(256) void qkv_proj_kernel(
    const bf16* __restrict__ X, const bf16* __restrict__ wq,
    const bf16* __restrict__ wk, const bf16* __restrict__ wv,
    bf16* __restrict__ Q, bf16* __restrict__ K, bf16* __restrict__ Vt) {
  __shared__ bf16 Bs[2][128 * LDB];

  const int tid  = threadIdx.x;
  const int wave = tid >> 5, lane = tid & 31;
  const int M_GRPS = 32;  // 256 M-tiles / 8 waves
  const int N_GRPS = 8;   // 1024 / 128
  const int blk  = blockIdx.x;                  // 3 * 32 * 8 = 768
  const int w    = blk / (M_GRPS * N_GRPS);
  const int rem  = blk % (M_GRPS * N_GRPS);
  const int mgrp = rem / N_GRPS;
  const int ngrp = rem % N_GRPS;
  const int mtile = mgrp * 8 + wave;
  const bf16* W = (w == 0) ? wq : (w == 1) ? wk : wv;
  const bf16* Arow = X + (size_t)(mtile * 16) * Dc;
  const int nBase = ngrp * 128;

  // cooperative B loader: thread -> row n = tid/2, kchunk = (tid&1)*16
  const int cn = tid >> 1;
  const int ck = (tid & 1) << 4;
  const bf16* Wrow = W + (size_t)(nBase + cn) * Dc + ck;

  copy16(&Bs[0][cn * LDB + ck], Wrow);          // preload k = 0
  __syncthreads();

  v8f acc[8] = {};
  int buf = 0;
  for (int k = 0; k < Dc; k += 32) {
    if (k + 32 < Dc)                            // fill next buffer (overlaps wmma)
      copy16(&Bs[buf ^ 1][cn * LDB + ck], Wrow + k + 32);
    v16bf a = load_A_frag(Arow + k, Dc);
#pragma unroll
    for (int j = 0; j < 8; ++j) {
      v16bf b = load_B_frag(&Bs[buf][(j * 16) * LDB], LDB);
      acc[j] = wmma_bf16(a, b, acc[j]);
    }
    __syncthreads();
    buf ^= 1;
  }

  bf16* dstQK = (w == 0) ? Q : K;
  const int nIn   = lane & 15;
  const int mBase = mtile * 16 + ((lane >> 4) << 3);
#pragma unroll
  for (int j = 0; j < 8; ++j) {
    const int ncol = nBase + j * 16 + nIn;
    const int h = ncol >> 6, d = ncol & (HDc - 1);
#pragma unroll
    for (int r = 0; r < 8; ++r) {
      const int mrow = mBase + r;
      const int b = mrow >> 11, t = mrow & (Tc - 1);
      const bf16 val = (bf16)acc[j][r];
      if (w == 2)
        Vt[((size_t)(b * Hc + h) * HDc + d) * Tc + t] = val;
      else
        dstQK[((size_t)(b * Hc + h) * Tc + t) * HDc + d] = val;
    }
  }
}

// ---------------------------------------------------------------------------
// Stage 2: flash attention. Block = 8 consecutive 16-query tiles of one (b,h);
// K/V tiles for the current 32 keys staged in LDS (double-buffered) and shared
// by all 8 waves. The block iterates to the causal bound of its LAST q-tile;
// waves past their own diagonal produce fully-masked tiles (p=0, alpha=1 =>
// exact no-op), keeping barriers uniform.
// Computes S^T = K_tile * Q_tile^T so the score C-layout (lane = query,
// vgprs = keys) maps 1:1 onto the A-fragment layout needed for P in the
// P*V WMMA -- no LDS / no cross-lane transpose for P.
// ---------------------------------------------------------------------------
__global__ __launch_bounds__(256) void attn_kernel(
    const bf16* __restrict__ Q, const bf16* __restrict__ K,
    const bf16* __restrict__ Vt, bf16* __restrict__ AO,
    const int* __restrict__ maskp) {
  __shared__ bf16 Ks[2][32 * LDK];  // [key][d], padded
  __shared__ bf16 Vs[2][64 * LDB];  // [d][key], padded

  const int tid  = threadIdx.x;
  const int wave = tid >> 5, lane = tid & 31;
  const int blk  = blockIdx.x;      // B*H*16 = 512
  const int qgrp = blk & 15;
  const int bh   = blk >> 4;
  const int h = bh & (Hc - 1), b = bh >> 4;
  const int qt = qgrp * 8 + wave;
  const int causal = *maskp;

  const bf16* Qbh = Q  + (size_t)bh * Tc * HDc;   // [T][hd]
  const bf16* Kbh = K  + (size_t)bh * Tc * HDc;   // [T][hd]
  const bf16* Vbh = Vt + (size_t)bh * HDc * Tc;   // [hd][T]

  const int qBase = qt * 16;
  const int myQ   = qBase + (lane & 15);  // this lane's query (softmax stats)

  // cooperative loader indices
  const int krow = tid >> 3, kcol = (tid & 7) << 3;  // K tile: 32 rows x 64
  const int vrow = tid >> 2, vcol = (tid & 3) << 3;  // V tile: 64 rows x 32

  v8f o0 = {}, o1 = {}, o2 = {}, o3 = {};
  float m_run = -3.0e38f;
  float l_run = 0.0f;

  // Q B-fragments (B[d][q] = Q[q][d]) are loop-invariant: preload from global.
  const v16bf bq0 = load_B_frag(Qbh + (size_t)qBase * HDc + 0,  HDc);
  const v16bf bq1 = load_B_frag(Qbh + (size_t)qBase * HDc + 32, HDc);

  const int kLastBlk = causal ? (qgrp * 8 + 7) * 16 : (Tc - 32);

  // preload keys [0,32)
  copy8(&Ks[0][krow * LDK + kcol], Kbh + (size_t)krow * HDc + kcol);
  copy8(&Vs[0][vrow * LDB + vcol], Vbh + (size_t)vrow * Tc + vcol);
  __syncthreads();

  int buf = 0;
  for (int k0 = 0; k0 <= kLastBlk; k0 += 32) {
    if (k0 + 32 <= kLastBlk) {  // fill next buffer (overlaps this step's math)
      copy8(&Ks[buf ^ 1][krow * LDK + kcol],
            Kbh + (size_t)(k0 + 32 + krow) * HDc + kcol);
      copy8(&Vs[buf ^ 1][vrow * LDB + vcol],
            Vbh + (size_t)vrow * Tc + (k0 + 32) + vcol);
    }

    // S^T tiles: st0 = keys [k0,k0+16), st1 = keys [k0+16,k0+32); N = queries
    v8f st0 = {}, st1 = {};
    {
      v16bf a00 = load_A_frag(&Ks[buf][0 * LDK + 0],  LDK);
      v16bf a10 = load_A_frag(&Ks[buf][16 * LDK + 0], LDK);
      st0 = wmma_bf16(a00, bq0, st0);
      st1 = wmma_bf16(a10, bq0, st1);
      v16bf a01 = load_A_frag(&Ks[buf][0 * LDK + 32],  LDK);
      v16bf a11 = load_A_frag(&Ks[buf][16 * LDK + 32], LDK);
      st0 = wmma_bf16(a01, bq1, st0);
      st1 = wmma_bf16(a11, bq1, st1);
    }

    // scale, causal mask, row(=query) max
    const float SC = 0.125f;  // 1/sqrt(64)
    float s0[8], s1[8];
    float mt = -3.0e38f;
    const int kb = k0 + ((lane >> 4) << 3);  // key idx of vgpr r in st0
#pragma unroll
    for (int r = 0; r < 8; ++r) {
      float v0 = st0[r] * SC;
      float v1 = st1[r] * SC;
      if (causal) {
        if (kb + r > myQ)      v0 = -3.0e38f;
        if (kb + 16 + r > myQ) v1 = -3.0e38f;
      }
      s0[r] = v0; s1[r] = v1;
      mt = fmaxf(mt, fmaxf(v0, v1));
    }
    mt = fmaxf(mt, __shfl_xor(mt, 16, 32));  // combine the two key halves
    const float m_new = fmaxf(m_run, mt);
    const float alpha = __expf(m_run - m_new);

    // p = exp(s - m_new); build P A-fragment in-place (lane-local relayout!)
    union { v16bf v; bf16 e[16]; } pf;
    float rs = 0.0f;
#pragma unroll
    for (int r = 0; r < 8; ++r) {
      const float p0 = __expf(s0[r] - m_new);
      const float p1 = __expf(s1[r] - m_new);
      rs += p0 + p1;
      pf.e[r]     = (bf16)p0;  // A half g=0,w=r  <-> key tile 0, C vgpr r
      pf.e[8 + r] = (bf16)p1;  // A half g=1,w=r  <-> key tile 1, C vgpr r
    }
    rs += __shfl_xor(rs, 16, 32);
    l_run = l_run * alpha + rs;
    m_run = m_new;

    // rescale O accumulators: O row M = r + (lane/16)*8; alpha lives in lane M
#pragma unroll
    for (int r = 0; r < 8; ++r) {
      const float aM = __shfl(alpha, r + ((lane >> 4) << 3), 32);
      o0[r] *= aM; o1[r] *= aM; o2[r] *= aM; o3[r] *= aM;
    }

    // O += P * V
    v16bf bv0 = load_B_frag(&Vs[buf][(0 * 16) * LDB], LDB);
    v16bf bv1 = load_B_frag(&Vs[buf][(1 * 16) * LDB], LDB);
    v16bf bv2 = load_B_frag(&Vs[buf][(2 * 16) * LDB], LDB);
    v16bf bv3 = load_B_frag(&Vs[buf][(3 * 16) * LDB], LDB);
    o0 = wmma_bf16(pf.v, bv0, o0);
    o1 = wmma_bf16(pf.v, bv1, o1);
    o2 = wmma_bf16(pf.v, bv2, o2);
    o3 = wmma_bf16(pf.v, bv3, o3);

    __syncthreads();
    buf ^= 1;
  }

  // normalize by row sums, store to attnOut [B][T][D] (bf16, D = H*hd)
  const int nIn = lane & 15;
#pragma unroll
  for (int r = 0; r < 8; ++r) {
    const float lM  = __shfl(l_run, r + ((lane >> 4) << 3), 32);
    const float inv = 1.0f / lM;
    const int t = qBase + r + ((lane >> 4) << 3);
    bf16* row = AO + (size_t)(b * Tc + t) * Dc + h * HDc;
    row[ 0 + nIn] = (bf16)(o0[r] * inv);
    row[16 + nIn] = (bf16)(o1[r] * inv);
    row[32 + nIn] = (bf16)(o2[r] * inv);
    row[48 + nIn] = (bf16)(o3[r] * inv);
  }
}

// ---------------------------------------------------------------------------
// Stage 3: output projection + bias.  out = attnOut @ wo^T + bo  (fp32 out)
// Same cooperative structure as stage 1.
// ---------------------------------------------------------------------------
__global__ __launch_bounds__(256) void out_proj_kernel(
    const bf16* __restrict__ AO, const bf16* __restrict__ wo,
    const float* __restrict__ bo, float* __restrict__ out) {
  __shared__ bf16 Bs[2][128 * LDB];

  const int tid  = threadIdx.x;
  const int wave = tid >> 5, lane = tid & 31;
  const int N_GRPS = 8;
  const int blk  = blockIdx.x;          // 32 * 8 = 256
  const int mgrp = blk / N_GRPS;
  const int ngrp = blk % N_GRPS;
  const int mtile = mgrp * 8 + wave;
  const int nBase = ngrp * 128;
  const bf16* Arow = AO + (size_t)(mtile * 16) * Dc;

  const int cn = tid >> 1;
  const int ck = (tid & 1) << 4;
  const bf16* Wrow = wo + (size_t)(nBase + cn) * Dc + ck;

  copy16(&Bs[0][cn * LDB + ck], Wrow);
  __syncthreads();

  v8f acc[8] = {};
  int buf = 0;
  for (int k = 0; k < Dc; k += 32) {
    if (k + 32 < Dc)
      copy16(&Bs[buf ^ 1][cn * LDB + ck], Wrow + k + 32);
    v16bf a = load_A_frag(Arow + k, Dc);
#pragma unroll
    for (int j = 0; j < 8; ++j) {
      v16bf b = load_B_frag(&Bs[buf][(j * 16) * LDB], LDB);
      acc[j] = wmma_bf16(a, b, acc[j]);
    }
    __syncthreads();
    buf ^= 1;
  }

  const int nIn   = lane & 15;
  const int mBase = mtile * 16 + ((lane >> 4) << 3);
  float bias[8];
#pragma unroll
  for (int j = 0; j < 8; ++j) bias[j] = bo[nBase + j * 16 + nIn];
#pragma unroll
  for (int j = 0; j < 8; ++j) {
#pragma unroll
    for (int r = 0; r < 8; ++r) {
      out[(size_t)(mBase + r) * Dc + nBase + j * 16 + nIn] = acc[j][r] + bias[j];
    }
  }
}

// ---------------------------------------------------------------------------
// Host launch
// ---------------------------------------------------------------------------
extern "C" void kernel_launch(void* const* d_in, const int* in_sizes, int n_in,
                              void* d_out, int out_size, void* d_ws, size_t ws_size,
                              hipStream_t stream) {
  (void)in_sizes; (void)n_in; (void)out_size; (void)ws_size;
  const float* x    = (const float*)d_in[0];
  const float* wqf  = (const float*)d_in[1];
  const float* wkf  = (const float*)d_in[2];
  const float* wvf  = (const float*)d_in[3];
  const float* wof  = (const float*)d_in[4];
  const float* bo   = (const float*)d_in[5];
  const int*   mask = (const int*)d_in[6];
  float* out = (float*)d_out;

  const size_t per  = (size_t)Bc * Hc * Tc * HDc;  // 4 Mi elems (Q/K/Vt/AO each)
  const size_t nX   = (size_t)Bc * Tc * Dc;        // 4 Mi elems
  const size_t nW   = (size_t)Dc * Dc;             // 1 Mi elems
  bf16* Q   = (bf16*)d_ws;
  bf16* K   = Q + per;
  bf16* V   = K + per;
  bf16* AO  = V + per;
  bf16* Xb  = AO + per;       // [B*T][D] bf16
  bf16* Wqb = Xb + nX;
  bf16* Wkb = Wqb + nW;
  bf16* Wvb = Wkb + nW;
  bf16* Wob = Wvb + nW;       // total ws use = 48 MB

  // Stage 0: one-pass fp32 -> bf16 conversions.
  cvt_bf16_kernel<<<dim3((int)(nX / 4 / 256)), dim3(256), 0, stream>>>(x,   Xb,  (int)(nX / 4));
  cvt_bf16_kernel<<<dim3((int)(nW / 4 / 256)), dim3(256), 0, stream>>>(wqf, Wqb, (int)(nW / 4));
  cvt_bf16_kernel<<<dim3((int)(nW / 4 / 256)), dim3(256), 0, stream>>>(wkf, Wkb, (int)(nW / 4));
  cvt_bf16_kernel<<<dim3((int)(nW / 4 / 256)), dim3(256), 0, stream>>>(wvf, Wvb, (int)(nW / 4));
  cvt_bf16_kernel<<<dim3((int)(nW / 4 / 256)), dim3(256), 0, stream>>>(wof, Wob, (int)(nW / 4));

  // Stage 1: 3 weights * 32 M-groups * 8 N-groups = 768 blocks (8 waves each)
  qkv_proj_kernel<<<dim3(768), dim3(256), 0, stream>>>(Xb, Wqb, Wkb, Wvb, Q, K, V);
  // Stage 2: B*H * 16 q-groups = 512 blocks (8 q-tiles each)
  attn_kernel<<<dim3(512), dim3(256), 0, stream>>>(Q, K, V, AO, mask);
  // Stage 3: 32 M-groups * 8 N-groups = 256 blocks
  out_proj_kernel<<<dim3(256), dim3(256), 0, stream>>>(AO, Wob, bo, out);
}
